// ODEFunc_7713761264213
// MI455X (gfx1250) — matrix operand order repaired and checked by reading, not compiled, not run on hardware
//
#include <hip/hip_runtime.h>
#include <hip/hip_bf16.h>
#include <math.h>

typedef _Float16 half_t;
typedef half_t v16h __attribute__((ext_vector_type(16)));
typedef half_t v8h  __attribute__((ext_vector_type(8)));
typedef float  v8f  __attribute__((ext_vector_type(8)));

#define HIDN   256
#define SDIM   137
#define KX     160   // state K padded to 5*32
#define KCAT   416   // 256 + 160
#define NCAT   144   // output cols padded to 9*16
#define BROWS  64    // batch rows per block
#define NTHR   256   // 8 waves
#define STRX   168   // LDS halfword stride for x tile
#define STRH   264   // LDS halfword stride for h tiles

__device__ __forceinline__ v16h cat8(v8h a, v8h b) {
  return __builtin_shufflevector(a, b, 0,1,2,3,4,5,6,7,8,9,10,11,12,13,14,15);
}
__device__ __forceinline__ v8f zero8() {
  v8f z;
#pragma unroll
  for (int i = 0; i < 8; ++i) z[i] = 0.f;
  return z;
}
__device__ __forceinline__ float fast_tanh(float x) {
  float e = __expf(2.0f * x);                       // v_exp_f32
  return (e - 1.0f) * __builtin_amdgcn_rcpf(e + 1.0f);
}
__device__ __forceinline__ v8f wmma_f16(v16h a, v16h b, v8f c) {
  return __builtin_amdgcn_wmma_f32_16x16x32_f16(false, a, false, b, (short)0, c, false, false);
}

// 32 rows (2 M-tiles) x NT*16 cols per wave; each B fragment feeds 2 WMMAs.
// A: row-major f16 in LDS (stride astr halfwords).  B: W row-major [N][Ks] f16 in global.
// acc layout: acc[0..NT-1] = M-tile 0, acc[NT..2NT-1] = M-tile 1.
template<int NT>
__device__ __forceinline__ void gemm_tile2(const half_t* __restrict__ As, int astr, int arow,
                                           const half_t* __restrict__ W, int Ks, int wcol0,
                                           int K, v8f* acc, int lane)
{
  const int m  = lane & 15;
  const int hi = lane >> 4;
  for (int kb = 0; kb < K; kb += 32) {
    // A 16x32 fragment: halves 0..7 = K[kb+8*hi ..), halves 8..15 = K[kb+16+8*hi ..)
    const half_t* ap0 = As + (arow + m) * astr + kb + hi * 8;
    const half_t* ap1 = ap0 + 16 * astr;
    v16h a0 = cat8(*(const v8h*)ap0, *(const v8h*)(ap0 + 16));
    v16h a1 = cat8(*(const v8h*)ap1, *(const v8h*)(ap1 + 16));
#pragma unroll
    for (int nt = 0; nt < NT; ++nt) {
      // B 32x16 fragment: lane col = m, halves 0..7 = K[kb+8*hi ..), halves 8..15 = K[kb+16+8*hi ..)
      const half_t* bp = W + (size_t)(wcol0 + nt * 16 + m) * Ks + kb + hi * 8;
      v16h b = cat8(*(const v8h*)bp, *(const v8h*)(bp + 16));
      acc[nt]      = wmma_f16(a0, b, acc[nt]);
      acc[NT + nt] = wmma_f16(a1, b, acc[NT + nt]);
    }
  }
}

// ---------------- prep kernels ----------------

__global__ void prep_misc(const float* t, const float* W0, const float* b0,
                          const float* bout,
                          const float* ta_out_w, const float* ta_out_b, const float* ta_in_b,
                          const float* lm_out_w, const float* lm_out_b, const float* lm_in_b,
                          float* b0e, float* bcat)
{
  int tid = threadIdx.x;
  float ang = t[0] * 0.26179938779914944f;  // 2*pi/24
  float sc = __sinf(ang), cc = __cosf(ang);
  if (tid < 256)
    b0e[tid] = b0[tid] + sc * W0[tid * 139 + 137] + cc * W0[tid * 139 + 138];
  if (tid < NCAT) {
    float v = (tid < SDIM) ? bout[tid] : 0.f;
    if (tid < 128) {
      float c = ta_out_b[tid];
      for (int i = 0; i < 128; ++i) c += ta_out_w[tid * 128 + i] * ta_in_b[256 + i];
      v += 0.1f * c;
    } else if (tid < 136) {
      int a = tid - 128;
      float c = lm_out_b[a];
      for (int i = 0; i < 8; ++i) c += lm_out_w[a * 8 + i] * lm_in_b[16 + i];
      v += 0.1f * c;
    }
    bcat[tid] = v;
  }
}

__global__ void prep_w0(const float* W0, half_t* W0h)
{
  int idx = blockIdx.x * 256 + threadIdx.x;   // 256*160
  int n = idx / KX, k = idx - n * KX;
  float v = (k < 139 - 2) ? W0[n * 139 + k] : 0.f;  // cols 0..136; tv folded into bias
  W0h[idx] = (half_t)v;
}

__global__ void prep_blk(const float* W1, const float* W2, half_t* W1h, half_t* W2h)
{
  int idx = blockIdx.x * 256 + threadIdx.x;   // 4*256*256
  W1h[idx] = (half_t)W1[idx];
  W2h[idx] = (half_t)W2[idx];
}

__global__ void prep_wcat(const float* Wout, const float* ta_out_w, const float* ta_in_w,
                          const float* lm_out_w, const float* lm_in_w, half_t* Wcat)
{
  int idx = blockIdx.x * 256 + threadIdx.x;   // 144*416
  int j = idx / KCAT, k = idx - j * KCAT;
  float v = 0.f;
  if (k < HIDN) {
    if (j < SDIM) v = Wout[j * HIDN + k];
  } else {
    int s = k - HIDN;  // state index
    if (j < 128 && s < 128) {
      float d = 0.f;
      for (int i = 0; i < 128; ++i)
        d += ta_out_w[j * 128 + i] * ta_in_w[(256 + i) * 128 + s];  // out_w @ Wv_ta
      v = 0.1f * d;
      if (s == j) v -= 0.1f;
    } else if (j >= 128 && j < 136 && s >= 128 && s < 136) {
      int a = j - 128, b = s - 128;
      float d = 0.f;
      for (int i = 0; i < 8; ++i)
        d += lm_out_w[a * 8 + i] * lm_in_w[(16 + i) * 8 + b];       // out_w @ Wv_lm
      v = 0.1f * d;
      if (s == j) v -= 0.1f;
    }
  }
  Wcat[idx] = (half_t)v;
}

// ---------------- main kernel ----------------

__global__ __launch_bounds__(NTHR) void ode_main(
    const float* __restrict__ state,
    const half_t* __restrict__ W0h, const float* __restrict__ b0e,
    const half_t* __restrict__ W1h, const float* __restrict__ b1,
    const half_t* __restrict__ W2h, const float* __restrict__ b2,
    const half_t* __restrict__ Wcat, const float* __restrict__ bcat,
    float* __restrict__ out)
{
  __shared__ __attribute__((aligned(16))) half_t xs[BROWS * STRX];
  __shared__ __attribute__((aligned(16))) half_t hA[BROWS * STRH];
  __shared__ __attribute__((aligned(16))) half_t hB[BROWS * STRH];

  const int tid   = threadIdx.x;
  const int lane  = tid & 31;
  const int wave  = tid >> 5;          // 8 waves
  const int mrow0 = (wave >> 2) * 32;  // 2 row-groups of 32 rows
  const int quad  = wave & 3;          // col quarter: quad*64 .. +64
  const int qcol0 = quad * 64;
  const size_t gRow0 = (size_t)blockIdx.x * BROWS;
  const int m  = lane & 15;
  const int hi = lane >> 4;

  // stage state tile as f16, zero-padded K 137->160
  for (int idx = tid; idx < BROWS * STRX; idx += NTHR) {
    int r = idx / STRX, c = idx - r * STRX;
    float v = (c < SDIM) ? state[(gRow0 + r) * SDIM + c] : 0.f;
    xs[idx] = (half_t)v;
  }
  __syncthreads();

  // GEMM0: h = relu(x @ W0h^T + b0e)
  {
    v8f acc[8];
#pragma unroll
    for (int i = 0; i < 8; ++i) acc[i] = zero8();
    gemm_tile2<4>(xs, STRX, mrow0, W0h, KX, qcol0, KX, acc, lane);
#pragma unroll
    for (int mt = 0; mt < 2; ++mt)
#pragma unroll
      for (int nt = 0; nt < 4; ++nt) {
        int ncol = qcol0 + nt * 16 + m;
        float bb = b0e[ncol];
#pragma unroll
        for (int r = 0; r < 8; ++r) {
          float v = acc[mt * 4 + nt][r] + bb;
          hA[(mrow0 + mt * 16 + hi * 8 + r) * STRH + ncol] = (half_t)(v > 0.f ? v : 0.f);
        }
      }
  }
  __syncthreads();

  // 4 residual blocks
  for (int blk = 0; blk < 4; ++blk) {
    const half_t* W1 = W1h + blk * HIDN * HIDN;
    const half_t* W2 = W2h + blk * HIDN * HIDN;
    const float*  bb1 = b1 + blk * HIDN;
    const float*  bb2 = b2 + blk * HIDN;

    v8f acc[8];
#pragma unroll
    for (int i = 0; i < 8; ++i) acc[i] = zero8();
    gemm_tile2<4>(hA, STRH, mrow0, W1, HIDN, qcol0, HIDN, acc, lane);
#pragma unroll
    for (int mt = 0; mt < 2; ++mt)
#pragma unroll
      for (int nt = 0; nt < 4; ++nt) {
        int ncol = qcol0 + nt * 16 + m;
        float bb = bb1[ncol];
#pragma unroll
        for (int r = 0; r < 8; ++r)
          hB[(mrow0 + mt * 16 + hi * 8 + r) * STRH + ncol] =
              (half_t)fast_tanh(acc[mt * 4 + nt][r] + bb);
      }
    __syncthreads();

    v8f acc2[8];
#pragma unroll
    for (int i = 0; i < 8; ++i) acc2[i] = zero8();
    gemm_tile2<4>(hB, STRH, mrow0, W2, HIDN, qcol0, HIDN, acc2, lane);
#pragma unroll
    for (int mt = 0; mt < 2; ++mt)
#pragma unroll
      for (int nt = 0; nt < 4; ++nt) {
        int ncol = qcol0 + nt * 16 + m;
        float bb = bb2[ncol];
#pragma unroll
        for (int r = 0; r < 8; ++r) {
          int rr = (mrow0 + mt * 16 + hi * 8 + r) * STRH + ncol;
          float hold = (float)hA[rr];
          hA[rr] = (half_t)fast_tanh(hold + acc2[mt * 4 + nt][r] + bb);
        }
      }
    __syncthreads();
  }

  // fused output: out = [h | x] @ Wcat^T + bcat   (delta terms folded in)
  // 9 N-tiles split 3/2/2/2 across the 4 col-quads.
  if (quad == 0) {
    v8f acc[6];
#pragma unroll
    for (int i = 0; i < 6; ++i) acc[i] = zero8();
    gemm_tile2<3>(hA, STRH, mrow0, Wcat, KCAT, 0, HIDN, acc, lane);
    gemm_tile2<3>(xs, STRX, mrow0, Wcat + HIDN, KCAT, 0, KX, acc, lane);
#pragma unroll
    for (int mt = 0; mt < 2; ++mt)
#pragma unroll
      for (int nt = 0; nt < 3; ++nt) {
        int ncol = nt * 16 + m;
        float bb = bcat[ncol];
#pragma unroll
        for (int r = 0; r < 8; ++r)
          out[(gRow0 + mrow0 + mt * 16 + hi * 8 + r) * SDIM + ncol] =
              acc[mt * 3 + nt][r] + bb;
      }
  } else {
    const int wc = (quad == 1) ? 48 : (quad == 2 ? 80 : 112);
    v8f acc[4];
#pragma unroll
    for (int i = 0; i < 4; ++i) acc[i] = zero8();
    gemm_tile2<2>(hA, STRH, mrow0, Wcat, KCAT, wc, HIDN, acc, lane);
    gemm_tile2<2>(xs, STRX, mrow0, Wcat + HIDN, KCAT, wc, KX, acc, lane);
#pragma unroll
    for (int mt = 0; mt < 2; ++mt)
#pragma unroll
      for (int nt = 0; nt < 2; ++nt) {
        int ncol = wc + nt * 16 + m;
        if (ncol < SDIM) {
          float bb = bcat[ncol];
#pragma unroll
          for (int r = 0; r < 8; ++r)
            out[(gRow0 + mrow0 + mt * 16 + hi * 8 + r) * SDIM + ncol] =
                acc[mt * 2 + nt][r] + bb;
        }
      }
  }
}

// ---------------- launcher ----------------

extern "C" void kernel_launch(void* const* d_in, const int* in_sizes, int n_in,
                              void* d_out, int out_size, void* d_ws, size_t ws_size,
                              hipStream_t stream) {
  const float* t       = (const float*)d_in[0];
  const float* state   = (const float*)d_in[1];
  const float* W0      = (const float*)d_in[2];
  const float* b0      = (const float*)d_in[3];
  const float* blkW1   = (const float*)d_in[4];
  const float* blkb1   = (const float*)d_in[5];
  const float* blkW2   = (const float*)d_in[6];
  const float* blkb2   = (const float*)d_in[7];
  const float* Wout    = (const float*)d_in[8];
  const float* bout    = (const float*)d_in[9];
  const float* lm_in_w = (const float*)d_in[10];
  const float* lm_in_b = (const float*)d_in[11];
  const float* lm_out_w= (const float*)d_in[12];
  const float* lm_out_b= (const float*)d_in[13];
  const float* ta_in_w = (const float*)d_in[14];
  const float* ta_in_b = (const float*)d_in[15];
  const float* ta_out_w= (const float*)d_in[16];
  const float* ta_out_b= (const float*)d_in[17];
  float* out = (float*)d_out;

  char* ws = (char*)d_ws;
  half_t* W0h  = (half_t*)(ws);                       // 256*160*2   = 81920
  half_t* W1h  = (half_t*)(ws + 81920);               // 4*65536*2   = 524288
  half_t* W2h  = (half_t*)(ws + 606208);              // 524288
  half_t* Wcat = (half_t*)(ws + 1130496);             // 144*416*2   = 119808
  float*  b0e  = (float*) (ws + 1250304);             // 1024
  float*  bcat = (float*) (ws + 1251328);             // 576

  prep_misc<<<1, 256, 0, stream>>>(t, W0, b0, bout, ta_out_w, ta_out_b, ta_in_b,
                                   lm_out_w, lm_out_b, lm_in_b, b0e, bcat);
  prep_w0  <<<(256 * KX) / 256, 256, 0, stream>>>(W0, W0h);
  prep_blk <<<(4 * HIDN * HIDN) / 256, 256, 0, stream>>>(blkW1, blkW2, W1h, W2h);
  prep_wcat<<<(NCAT * KCAT) / 256, 256, 0, stream>>>(Wout, ta_out_w, ta_in_w,
                                                     lm_out_w, lm_in_w, Wcat);

  int B = in_sizes[1] / SDIM;
  ode_main<<<B / BROWS, NTHR, 0, stream>>>(state, W0h, b0e, W1h, blkb1, W2h, blkb2,
                                           Wcat, bcat, out);
}
